// MSAPairWeightedAveraging_53833120088456
// MI455X (gfx1250) — matrix operand order repaired and checked by reading, not compiled, not run on hardware
//
#include <hip/hip_runtime.h>
#include <hip/hip_bf16.h>

#define Ssz 512
#define Nsz 384
#define DMsz 64
#define DPsz 128
#define Hsz 8
#define DHsz 32
#define DIsz 256

typedef __bf16 v16bf __attribute__((ext_vector_type(16)));
typedef float  v8f   __attribute__((ext_vector_type(8)));

union U16 { float4 f4[2]; v16bf v; };

// A-tile: elements 0..7 at p[0..7], elements 8..15 at p[16..23] (K split per lane-half)
__device__ __forceinline__ v16bf load_a16(const __bf16* p) {
    U16 u;
    u.f4[0] = *reinterpret_cast<const float4*>(p);
    u.f4[1] = *reinterpret_cast<const float4*>(p + 16);
    return u.v;
}
// B-tile: elements 0..7 at p[0..7], elements 8..15 at p[16..23] (K split per lane-half)
__device__ __forceinline__ v16bf load_b16(const __bf16* p) {
    U16 u;
    u.f4[0] = *reinterpret_cast<const float4*>(p);
    u.f4[1] = *reinterpret_cast<const float4*>(p + 16);
    return u.v;
}

__device__ __forceinline__ v8f wmma_bf16(v16bf a, v16bf b, v8f c) {
    return __builtin_amdgcn_wmma_f32_16x16x32_bf16(false, a, false, b, (short)0, c, false, false);
}

// ---------------- Kernel 0: convert + transpose weight matrices to bf16 ----------------
__global__ __launch_bounds__(256) void k_prep(const float* __restrict__ Wvg,
                                              const float* __restrict__ Wout,
                                              __bf16* __restrict__ WvgT,
                                              __bf16* __restrict__ WoutT) {
    int t = blockIdx.x * blockDim.x + threadIdx.x;
    if (t < DMsz * 2 * DIsz) {                 // WvgT[c][k] = Wvg[k][c], 512x64
        int c = t / DMsz, k = t % DMsz;
        WvgT[t] = (__bf16)Wvg[k * (2 * DIsz) + c];
    } else {
        int t2 = t - DMsz * 2 * DIsz;
        if (t2 < DMsz * DIsz) {                // WoutT[n][k] = Wout[k][n], 64x256
            int n = t2 / DIsz, k = t2 % DIsz;
            WoutT[t2] = (__bf16)Wout[k * DMsz + n];
        }
    }
}

// ---------------- Kernel 1: pairwise LN -> per-head bias -> softmax -> bf16 weights ----
__global__ __launch_bounds__(256) void k_bias_softmax(const float* __restrict__ pw,
                                                      const unsigned char* __restrict__ mask,
                                                      const float* __restrict__ g2,
                                                      const float* __restrict__ b2,
                                                      const float* __restrict__ Wb,
                                                      __bf16* __restrict__ Wbf) {
    int i = blockIdx.x;
    int lane = threadIdx.x & 31;
    int wave = threadIdx.x >> 5;
    __shared__ float sbias[Hsz][Nsz];

    for (int j = wave; j < Nsz; j += 8) {
        const float* row = pw + (size_t)(i * Nsz + j) * DPsz;
        float x[4];
        float s = 0.f;
#pragma unroll
        for (int k = 0; k < 4; k++) { x[k] = row[lane + 32 * k]; s += x[k]; }
#pragma unroll
        for (int o = 16; o >= 1; o >>= 1) s += __shfl_xor(s, o, 32);
        float m = s * (1.0f / DPsz);
        float v = 0.f;
#pragma unroll
        for (int k = 0; k < 4; k++) { float d = x[k] - m; v += d * d; }
#pragma unroll
        for (int o = 16; o >= 1; o >>= 1) v += __shfl_xor(v, o, 32);
        float inv = rsqrtf(v * (1.0f / DPsz) + 1e-5f);
        float y[4];
#pragma unroll
        for (int k = 0; k < 4; k++) {
            int p = lane + 32 * k;
            y[k] = (x[k] - m) * inv * g2[p] + b2[p];
        }
        float acc[8];
#pragma unroll
        for (int h = 0; h < 8; h++) {
            acc[h] = 0.f;
#pragma unroll
            for (int k = 0; k < 4; k++) acc[h] += y[k] * Wb[(lane + 32 * k) * Hsz + h];
        }
#pragma unroll
        for (int h = 0; h < 8; h++) {
#pragma unroll
            for (int o = 16; o >= 1; o >>= 1) acc[h] += __shfl_xor(acc[h], o, 32);
        }
        if (lane == 0) {
#pragma unroll
            for (int h = 0; h < 8; h++) sbias[h][j] = acc[h];
        }
    }
    __syncthreads();

    int h = wave;                       // 8 waves, one head each
    float vals[12];
    float mx = -__builtin_inff();
#pragma unroll
    for (int k = 0; k < 12; k++) {
        int j = lane + 32 * k;
        float t = mask[j] ? sbias[h][j] : -__builtin_inff();
        vals[k] = t;
        mx = fmaxf(mx, t);
    }
#pragma unroll
    for (int o = 16; o >= 1; o >>= 1) mx = fmaxf(mx, __shfl_xor(mx, o, 32));
    float se = 0.f;
#pragma unroll
    for (int k = 0; k < 12; k++) { float e = __expf(vals[k] - mx); vals[k] = e; se += e; }
#pragma unroll
    for (int o = 16; o >= 1; o >>= 1) se += __shfl_xor(se, o, 32);
    float rs = 1.0f / se;
#pragma unroll
    for (int k = 0; k < 12; k++) {
        int j = lane + 32 * k;
        Wbf[((size_t)h * Nsz + i) * Nsz + j] = (__bf16)(vals[k] * rs);
    }
}

// ---------------- Kernel 2: LN(msa) @ W_vg -> values (transposed) + sigmoid gates ------
__global__ __launch_bounds__(64) void k_vg(const float* __restrict__ msa,
                                           const float* __restrict__ g1,
                                           const float* __restrict__ b1,
                                           const __bf16* __restrict__ WvgT,
                                           __bf16* __restrict__ Vbf,
                                           __bf16* __restrict__ Gbf) {
    __shared__ __bf16 xln[16][DMsz];
    int r0 = blockIdx.x * 16;            // 16 (s,n) rows per block; tile stays in one s
    int s  = r0 / Nsz;
    int n0 = r0 % Nsz;
    int t  = threadIdx.x;
    int rl = t >> 2;                     // row 0..15
    int seg = t & 3;                     // 16-col segment
    const float* row = msa + (size_t)(r0 + rl) * DMsz;
    float x[16];
    float ssum = 0.f;
#pragma unroll
    for (int k = 0; k < 16; k++) { x[k] = row[seg * 16 + k]; ssum += x[k]; }
    ssum += __shfl_xor(ssum, 1, 32);
    ssum += __shfl_xor(ssum, 2, 32);
    float m = ssum * (1.0f / DMsz);
    float var = 0.f;
#pragma unroll
    for (int k = 0; k < 16; k++) { float d = x[k] - m; var += d * d; }
    var += __shfl_xor(var, 1, 32);
    var += __shfl_xor(var, 2, 32);
    float inv = rsqrtf(var * (1.0f / DMsz) + 1e-5f);
#pragma unroll
    for (int k = 0; k < 16; k++) {
        int c = seg * 16 + k;
        xln[rl][c] = (__bf16)((x[k] - m) * inv * g1[c] + b1[c]);
    }
    __syncthreads();

    int wave = t >> 5, lane = t & 31, lanelo = lane & 15, hi = lane >> 4;
    const __bf16* arow = &xln[lanelo][0];
    v16bf a0 = load_a16(arow + 8 * hi);        // K 0..31
    v16bf a1 = load_a16(arow + 32 + 8 * hi);   // K 32..63

    for (int nt = 0; nt < 16; ++nt) {
        int c0 = wave * 256 + nt * 16;         // wave0: values cols, wave1: gates cols
        const __bf16* bp = WvgT + (size_t)(c0 + lanelo) * DMsz;
        v8f acc = {};
        acc = wmma_bf16(a0, load_b16(bp + 8 * hi), acc);
        acc = wmma_bf16(a1, load_b16(bp + 32 + 8 * hi), acc);
        int col = c0 + lanelo;
        if (wave == 0) {
            int h = col >> 5, d = col & 31;
            __bf16* dst = Vbf + (((size_t)h * Ssz + s) * DHsz + d) * Nsz;
#pragma unroll
            for (int f = 0; f < 8; f++) {
                int n = n0 + f + 8 * hi;
                dst[n] = (__bf16)acc[f];
            }
        } else {
            int cg = col - 256;
            int h = cg >> 5, d = cg & 31;
            __bf16* dst = Gbf + ((size_t)h * Ssz + s) * Nsz * DHsz + d;
#pragma unroll
            for (int f = 0; f < 8; f++) {
                int n = n0 + f + 8 * hi;
                dst[(size_t)n * DHsz] = (__bf16)(1.0f / (1.0f + __expf(-acc[f])));
            }
        }
    }
}

// ---------------- Kernel 3: out[h,s,i,d] = sum_j W[h,i,j] * V[h,s,j,d]; gate; pack ctx -
// Register-blocked: 2 i-tiles x 2 s per block (8 accumulators, shared A/B fragments)
__global__ __launch_bounds__(32) void k_pwa(const __bf16* __restrict__ Wbf,
                                            const __bf16* __restrict__ Vbf,
                                            const __bf16* __restrict__ Gbf,
                                            __bf16* __restrict__ Ctx) {
    int it = blockIdx.x;                 // i-super-tile 0..11 (32 rows of i)
    int s0 = blockIdx.y * 2;             // two consecutive s
    int h  = blockIdx.z;
    int lane = threadIdx.x & 31, lanelo = lane & 15, hi = lane >> 4;
    int i0 = it * 32;

    const __bf16* Wrow0 = Wbf + ((size_t)h * Nsz + (i0 + lanelo)) * Nsz;
    const __bf16* Wrow1 = Wrow0 + (size_t)16 * Nsz;
    const __bf16* Vbase = Vbf + ((size_t)h * Ssz + s0) * DHsz * Nsz;   // [s][d][n]

    v8f acc[2][2][2] = {};               // [s][i-tile][d-tile]
    for (int j0 = 0; j0 < Nsz; j0 += 32) {
        v16bf a0 = load_a16(Wrow0 + j0 + 8 * hi);
        v16bf a1 = load_a16(Wrow1 + j0 + 8 * hi);
#pragma unroll
        for (int ss = 0; ss < 2; ss++) {
            const __bf16* Vb = Vbase + (size_t)ss * (DHsz * Nsz);
            const __bf16* bp0 = Vb + (size_t)(lanelo)      * Nsz + j0 + 8 * hi;
            const __bf16* bp1 = Vb + (size_t)(16 + lanelo) * Nsz + j0 + 8 * hi;
            // prefetch next j-block of V (first touch streams from HBM)
            __builtin_prefetch((const void*)(bp0 + 32), 0, 3);
            v16bf b0 = load_b16(bp0);
            v16bf b1 = load_b16(bp1);
            acc[ss][0][0] = wmma_bf16(a0, b0, acc[ss][0][0]);
            acc[ss][0][1] = wmma_bf16(a0, b1, acc[ss][0][1]);
            acc[ss][1][0] = wmma_bf16(a1, b0, acc[ss][1][0]);
            acc[ss][1][1] = wmma_bf16(a1, b1, acc[ss][1][1]);
        }
    }
#pragma unroll
    for (int ss = 0; ss < 2; ss++) {
        int s = s0 + ss;
        const __bf16* Grow = Gbf + ((size_t)h * Ssz + s) * Nsz * DHsz;
#pragma unroll
        for (int ii = 0; ii < 2; ii++) {
#pragma unroll
            for (int f = 0; f < 8; f++) {
                int i = i0 + ii * 16 + f + 8 * hi;
                float gv0 = (float)Grow[(size_t)i * DHsz + lanelo];
                float gv1 = (float)Grow[(size_t)i * DHsz + 16 + lanelo];
                size_t cbase = ((size_t)s * Nsz + i) * DIsz + (size_t)h * DHsz;
                Ctx[cbase + lanelo]      = (__bf16)(acc[ss][ii][0][f] * gv0);
                Ctx[cbase + 16 + lanelo] = (__bf16)(acc[ss][ii][1][f] * gv1);
            }
        }
    }
}

// ---------------- Kernel 4: output projection ctx @ W_out -> fp32 out ------------------
// Register-blocked: 2 row-tiles per block share the 4 W_out B-tiles
__global__ __launch_bounds__(32) void k_proj(const __bf16* __restrict__ Ctx,
                                             const __bf16* __restrict__ WoutT,
                                             float* __restrict__ out) {
    int r0 = blockIdx.x * 32;
    int lane = threadIdx.x & 31, lanelo = lane & 15, hi = lane >> 4;
    const __bf16* Ar0 = Ctx + (size_t)(r0 + lanelo) * DIsz;
    const __bf16* Ar1 = Ar0 + (size_t)16 * DIsz;
    v8f acc[2][4] = {};
    for (int k0 = 0; k0 < DIsz; k0 += 32) {
        v16bf a0 = load_a16(Ar0 + k0 + 8 * hi);
        v16bf a1 = load_a16(Ar1 + k0 + 8 * hi);
#pragma unroll
        for (int nt = 0; nt < 4; nt++) {
            v16bf b = load_b16(WoutT + (size_t)(nt * 16 + lanelo) * DIsz + k0 + 8 * hi);
            acc[0][nt] = wmma_bf16(a0, b, acc[0][nt]);
            acc[1][nt] = wmma_bf16(a1, b, acc[1][nt]);
        }
    }
#pragma unroll
    for (int rr = 0; rr < 2; rr++) {
#pragma unroll
        for (int nt = 0; nt < 4; nt++) {
#pragma unroll
            for (int f = 0; f < 8; f++) {
                int r = r0 + rr * 16 + f + 8 * hi;
                out[(size_t)r * DMsz + nt * 16 + lanelo] = acc[rr][nt][f];
            }
        }
    }
}

extern "C" void kernel_launch(void* const* d_in, const int* in_sizes, int n_in,
                              void* d_out, int out_size, void* d_ws, size_t ws_size,
                              hipStream_t stream) {
    const float* msa  = (const float*)d_in[0];
    const float* pw   = (const float*)d_in[1];
    const unsigned char* mask = (const unsigned char*)d_in[2];
    const float* ln1g = (const float*)d_in[3];
    const float* ln1b = (const float*)d_in[4];
    const float* Wvg  = (const float*)d_in[5];
    const float* ln2g = (const float*)d_in[6];
    const float* ln2b = (const float*)d_in[7];
    const float* Wb   = (const float*)d_in[8];
    const float* Wout = (const float*)d_in[9];
    float* out = (float*)d_out;

    char* ws = (char*)d_ws;
    const size_t szVG   = 100663296ULL;                 // 8*512*32*384 * 2B
    __bf16* WvgT  = (__bf16*)(ws);                      // 65536 B
    __bf16* WoutT = (__bf16*)(ws + 65536);              // 32768 B
    __bf16* Wbf   = (__bf16*)(ws + 98304);              // 2359296 B
    __bf16* Vbf   = (__bf16*)(ws + 2457600);            // szVG
    __bf16* Gbf   = (__bf16*)(ws + 2457600 + szVG);     // szVG
    __bf16* Ctx   = (__bf16*)(ws + 2457600 + 2 * szVG); // szVG

    k_prep<<<192, 256, 0, stream>>>(Wvg, Wout, WvgT, WoutT);
    k_bias_softmax<<<Nsz, 256, 0, stream>>>(pw, mask, ln2g, ln2b, Wb, Wbf);
    k_vg<<<(Ssz * Nsz) / 16, 64, 0, stream>>>(msa, ln1g, ln1b, WvgT, Vbf, Gbf);
    dim3 g3(Nsz / 32, Ssz / 2, Hsz);
    k_pwa<<<g3, 32, 0, stream>>>(Wbf, Vbf, Gbf, Ctx);
    k_proj<<<(Ssz * Nsz) / 32, 32, 0, stream>>>(Ctx, WoutT, out);
}
